// MixtralSparseMoeBlock_28621662061196
// MI455X (gfx1250) — matrix operand order repaired and checked by reading, not compiled, not run on hardware
//
#include <hip/hip_runtime.h>
#include <hip/hip_bf16.h>

// ---------------- problem constants (match reference) ----------------
#define NUM_EXPERTS 8
#define TOPK 2
#define HID 1024            // H
#define INTER 3584          // I
#define NTOK 2048           // T = B*S
#define TILE_M 16           // tokens per block tile
#define F_T 128             // inter-dim chunk per block iteration
#define STAGE_ROWS 4        // x rows async-staged per round (16KB fp32)

typedef __attribute__((ext_vector_type(16))) __bf16 bf16x16;
typedef __attribute__((ext_vector_type(8)))  __bf16 bf16x8;
typedef __attribute__((ext_vector_type(8)))  float  f32x8;

// ---------------------------------------------------------------------
// WMMA wrapper: D = A(16x32 bf16) x B(32x16 bf16) + C(16x16 f32)
// ---------------------------------------------------------------------
__device__ __forceinline__ f32x8 wmma_bf16(bf16x16 a, bf16x16 b, f32x8 c) {
  return __builtin_amdgcn_wmma_f32_16x16x32_bf16(
      /*neg_a=*/false, a, /*neg_b=*/false, b,
      /*c_mod=*/(short)0, c, /*reuse_a=*/false, /*reuse_b=*/false);
}

// A-fragment from bf16 LDS tile [TILE_M][pitch], top-left at column kbase.
// 16-bit A 16x32 layout (matches B): lane<16 -> M=lane, K = kbase+0..15
//                                    lane>=16 -> M=lane-16, K = kbase+16..31
__device__ __forceinline__ bf16x16 load_a_lds(const __bf16* base, int pitch,
                                              int mrow, int hs, int kbase) {
  const bf16x8 lo = *(const bf16x8*)(base + mrow * pitch + kbase + 2 * hs);
  const bf16x8 hi = *(const bf16x8*)(base + mrow * pitch + kbase + 2 * hs + 8);
  return __builtin_shufflevector(lo, hi, 0, 1, 2, 3, 4, 5, 6, 7,
                                 8, 9, 10, 11, 12, 13, 14, 15);
}

// B-fragment from global fp32 weights (row-major [N][K], K contiguous).
// 16-bit B 32x16 layout used here (same K order as A): lane<16 -> N=lane,  K = kbase+0..15
//                            lane>=16 -> N=lane-16, K = kbase+16..31
// Caller passes the per-lane row pointer already offset by kbase + ksel*16.
// fp32->bf16 conversions pack into v_cvt_pk_bf16_f32 and co-execute with WMMA.
__device__ __forceinline__ bf16x16 load_b_f32(const float* rowptr) {
  const float4* p = (const float4*)rowptr;
  bf16x16 b;
#pragma unroll
  for (int q = 0; q < 4; ++q) {
    float4 v = p[q];
    b[4 * q + 0] = (__bf16)v.x;
    b[4 * q + 1] = (__bf16)v.y;
    b[4 * q + 2] = (__bf16)v.z;
    b[4 * q + 3] = (__bf16)v.w;
  }
  return b;
}

// ---------------------------------------------------------------------
// Kernel 1: router. One wave per token: logits -> softmax -> top-2 ->
// renormalize. Writes per-token slot experts + gates.
// ---------------------------------------------------------------------
__global__ __launch_bounds__(256) void router_kernel(
    const float* __restrict__ x, const float* __restrict__ gw,
    int* __restrict__ tok_e, float* __restrict__ tok_g) {
  const int wave = threadIdx.x >> 5;
  const int lane = threadIdx.x & 31;
  const int t = blockIdx.x * 8 + wave;
  if (t >= NTOK) return;
  const float* xr = x + (size_t)t * HID;

  float acc[NUM_EXPERTS];
#pragma unroll
  for (int e = 0; e < NUM_EXPERTS; ++e) acc[e] = 0.f;
  for (int h = lane; h < HID; h += 32) {
    const float xv = xr[h];
#pragma unroll
    for (int e = 0; e < NUM_EXPERTS; ++e) acc[e] += xv * gw[e * HID + h];
  }
#pragma unroll
  for (int e = 0; e < NUM_EXPERTS; ++e) {
#pragma unroll
    for (int off = 16; off >= 1; off >>= 1)
      acc[e] += __shfl_xor(acc[e], off, 32);
  }
  if (lane == 0) {
    float mx = acc[0];
#pragma unroll
    for (int e = 1; e < NUM_EXPERTS; ++e) mx = fmaxf(mx, acc[e]);
    float p[NUM_EXPERTS], sum = 0.f;
#pragma unroll
    for (int e = 0; e < NUM_EXPERTS; ++e) { p[e] = __expf(acc[e] - mx); sum += p[e]; }
    const float inv = __builtin_amdgcn_rcpf(sum);
#pragma unroll
    for (int e = 0; e < NUM_EXPERTS; ++e) p[e] *= inv;
    int i0 = 0;
#pragma unroll
    for (int e = 1; e < NUM_EXPERTS; ++e) if (p[e] > p[i0]) i0 = e;
    int i1 = (i0 == 0) ? 1 : 0;
#pragma unroll
    for (int e = 0; e < NUM_EXPERTS; ++e)
      if (e != i0 && p[e] > p[i1]) i1 = e;
    const float r = 1.f / (p[i0] + p[i1]);  // keep renorm accurate
    tok_e[t * 2 + 0] = i0;
    tok_e[t * 2 + 1] = i1;
    tok_g[t * 2 + 0] = p[i0] * r;
    tok_g[t * 2 + 1] = p[i1] * r;
  }
}

// ---------------------------------------------------------------------
// Kernel 2: deterministic list build. Thread (e,slot) serially scans all
// tokens in order -> token-ordered lists, bitwise-stable across calls.
// ---------------------------------------------------------------------
__global__ void build_lists_kernel(const int* __restrict__ tok_e,
                                   const float* __restrict__ tok_g,
                                   int* __restrict__ counts,
                                   int* __restrict__ ltok,
                                   float* __restrict__ lgate) {
  const int id = threadIdx.x;  // 0..15 = e*2 + slot
  if (id >= NUM_EXPERTS * TOPK) return;
  const int e = id >> 1, k = id & 1;
  int c = 0;
  for (int t = 0; t < NTOK; ++t) {
    if (tok_e[t * 2 + k] == e) {
      ltok[(size_t)id * NTOK + c] = t;
      lgate[(size_t)id * NTOK + c] = tok_g[t * 2 + k];
      ++c;
    }
  }
  counts[id] = c;
}

// ---------------------------------------------------------------------
// Kernel 3: fused SwiGLU expert GEMM over a 16-token tile of one
// (expert, slot) list. 8 waves; wave w owns hid cols [16w,16w+16) in
// phase 1 and out cols [128w,128w+128) in phase 2. SLOT==0 writes out,
// SLOT==1 accumulates (deterministic: each token appears exactly once
// per slot pass). Gathered x rows are staged LDS-direct via
// GLOBAL_LOAD_ASYNC_TO_LDS_B128 (ASYNCcnt) then converted to bf16.
// ---------------------------------------------------------------------
template <int SLOT>
__global__ __launch_bounds__(256) void moe_expert_kernel(
    const float* __restrict__ x, const float* __restrict__ w1,
    const float* __restrict__ w2, const float* __restrict__ w3,
    const int* __restrict__ counts, const int* __restrict__ ltok,
    const float* __restrict__ lgate, float* __restrict__ out) {
  constexpr int XP = HID + 8;  // bf16 pitch: stride 2064B -> conflict-free b128
  constexpr int HP = F_T + 8;  // bf16 pitch: stride 272B  -> conflict-free b128

  __shared__ __align__(16) float xf_stage[STAGE_ROWS * HID];  // fp32 async landing pad
  __shared__ __align__(16) __bf16 x_lds[TILE_M * XP];
  __shared__ __align__(16) __bf16 hid_lds[TILE_M * HP];
  __shared__ int s_tok[TILE_M];
  __shared__ float s_gate[TILE_M];

  const int e = blockIdx.x;
  const int tile = blockIdx.y;
  const int id = e * 2 + SLOT;
  const int cnt = counts[id];
  const int row0 = tile * TILE_M;
  if (row0 >= cnt) return;
  const int rem = min(TILE_M, cnt - row0);

  const int tid = threadIdx.x;
  const int lane = tid & 31;
  const int wv = tid >> 5;
  const int mrow = lane & 15;       // lane's M (A) / N (B,C) sub-index
  const int ksel = lane >> 4;       // K-half select
  const int hs = ksel * 8;          // A fragment K sub-offset
  const int rowoff = ksel * 8;      // C fragment row offset

  if (tid < TILE_M) {
    const int r = (tid < rem) ? tid : 0;
    s_tok[tid] = ltok[(size_t)id * NTOK + row0 + r];
    s_gate[tid] = (tid < rem) ? lgate[(size_t)id * NTOK + row0 + tid] : 0.f;
  }
  __syncthreads();

  // ---- Stage gathered x rows: async global->LDS (fp32), then cvt to bf16.
  // STAGE_ROWS rows/round: STAGE_ROWS*HID*4B / 16B = 1024 b128 ops = 4/thread.
  const unsigned lds_base = (unsigned)(uintptr_t)xf_stage;  // low 32b = LDS addr
  for (int r0 = 0; r0 < TILE_M; r0 += STAGE_ROWS) {
#pragma unroll
    for (int j = 0; j < 4; ++j) {
      const int c = tid + 256 * j;            // 0..1023
      const int m = c >> 8;                   // 0..STAGE_ROWS-1
      const int bo = (c & 255) * 16;          // byte offset within row
      const unsigned lds_dst = lds_base + (unsigned)(m * HID * 4 + bo);
      const unsigned goff = (unsigned)s_tok[r0 + m] * (HID * 4) + (unsigned)bo;
      asm volatile("global_load_async_to_lds_b128 %0, %1, %2"
                   :: "v"(lds_dst), "v"(goff), "s"(x)
                   : "memory");
    }
    asm volatile("s_wait_asynccnt 0x0" ::: "memory");
    __syncthreads();
    for (int u = tid; u < STAGE_ROWS * HID; u += 256) {
      const int m = r0 + (u >> 10);           // HID == 1024
      const int h = u & (HID - 1);
      x_lds[m * XP + h] = (__bf16)xf_stage[u];
    }
    __syncthreads();
  }

  const f32x8 zero = {0.f, 0.f, 0.f, 0.f, 0.f, 0.f, 0.f, 0.f};
  f32x8 acc2[8];
#pragma unroll
  for (int s = 0; s < 8; ++s) acc2[s] = zero;

  const size_t ew13 = (size_t)e * INTER;  // expert row base for w1/w3
  const size_t ew2 = (size_t)e * HID;     // expert row base for w2

  for (int fb = 0; fb < INTER; fb += F_T) {
    // ---------- phase 1: hid tile [16 x F_T], wave w -> cols 16w..16w+16
    const float* w1r = w1 + (ew13 + fb + wv * 16 + mrow) * HID + ksel * 16;
    const float* w3r = w3 + (ew13 + fb + wv * 16 + mrow) * HID + ksel * 16;
    if (fb + F_T < INTER) {  // warm next chunk (global_prefetch_b8)
      __builtin_prefetch(w1r + (size_t)F_T * HID, 0, 0);
      __builtin_prefetch(w3r + (size_t)F_T * HID, 0, 0);
    }
    f32x8 a1 = zero, a3 = zero;
    for (int kk = 0; kk < HID; kk += 32) {
      const bf16x16 A = load_a_lds(x_lds, XP, mrow, hs, kk);
      const bf16x16 B1 = load_b_f32(w1r + kk);
      a1 = wmma_bf16(A, B1, a1);
      const bf16x16 B3 = load_b_f32(w3r + kk);
      a3 = wmma_bf16(A, B3, a3);
    }
    // silu(x@w1) * (x@w3) -> bf16 hid in LDS (C layout: elem i = row i+rowoff)
#pragma unroll
    for (int i = 0; i < 8; ++i) {
      const float z1 = a1[i];
      const float sig = __builtin_amdgcn_rcpf(1.f + __expf(-z1));  // v_rcp_f32
      hid_lds[(i + rowoff) * HP + wv * 16 + mrow] = (__bf16)(z1 * sig * a3[i]);
    }
    __syncthreads();

    // ---------- phase 2: out slice [16 x 128] per wave, K = this chunk
    for (int kk = 0; kk < F_T; kk += 32) {
      const bf16x16 A = load_a_lds(hid_lds, HP, mrow, hs, kk);
#pragma unroll
      for (int s = 0; s < 8; ++s) {
        const float* w2r = w2 + (ew2 + wv * 128 + s * 16 + mrow) * INTER +
                           fb + kk + ksel * 16;
        const bf16x16 B = load_b_f32(w2r);
        acc2[s] = wmma_bf16(A, B, acc2[s]);
      }
    }
    __syncthreads();  // hid_lds rewritten next iteration
  }

  // Epilogue: gate scale + (write | accumulate) to out. Row-guarded.
  float g[8];
#pragma unroll
  for (int i = 0; i < 8; ++i) g[i] = s_gate[i + rowoff];
#pragma unroll
  for (int s = 0; s < 8; ++s) {
    const int n = wv * 128 + s * 16 + mrow;
#pragma unroll
    for (int i = 0; i < 8; ++i) {
      const int m = i + rowoff;
      if (m < rem) {
        const size_t off = (size_t)s_tok[m] * HID + n;
        const float v = acc2[s][i] * g[i];
        if (SLOT == 0)
          out[off] = v;
        else
          out[off] += v;
      }
    }
  }
}

// ---------------------------------------------------------------------
extern "C" void kernel_launch(void* const* d_in, const int* in_sizes, int n_in,
                              void* d_out, int out_size, void* d_ws,
                              size_t ws_size, hipStream_t stream) {
  const float* x = (const float*)d_in[0];   // [1, 2048, 1024]
  const float* gw = (const float*)d_in[1];  // [8, 1024]
  const float* w1 = (const float*)d_in[2];  // [8, 3584, 1024]
  const float* w2 = (const float*)d_in[3];  // [8, 1024, 3584]
  const float* w3 = (const float*)d_in[4];  // [8, 3584, 1024]
  float* out = (float*)d_out;               // [1, 2048, 1024]

  // workspace carve-up (~288 KB total)
  char* ws = (char*)d_ws;
  int* tok_e = (int*)ws;      ws += (size_t)NTOK * 2 * sizeof(int);
  float* tok_g = (float*)ws;  ws += (size_t)NTOK * 2 * sizeof(float);
  int* counts = (int*)ws;     ws += 64 * sizeof(int);
  int* ltok = (int*)ws;       ws += (size_t)NUM_EXPERTS * TOPK * NTOK * sizeof(int);
  float* lgate = (float*)ws;

  router_kernel<<<NTOK / 8, 256, 0, stream>>>(x, gw, tok_e, tok_g);
  build_lists_kernel<<<1, 16, 0, stream>>>(tok_e, tok_g, counts, ltok, lgate);

  dim3 grid(NUM_EXPERTS, NTOK / TILE_M);
  moe_expert_kernel<0><<<grid, 256, 0, stream>>>(x, w1, w2, w3, counts, ltok,
                                                 lgate, out);
  moe_expert_kernel<1><<<grid, 256, 0, stream>>>(x, w1, w2, w3, counts, ltok,
                                                 lgate, out);
}
